// GumbelVectorQuantizer_80788334838455
// MI455X (gfx1250) — matrix operand order repaired and checked by reading, not compiled, not run on hardware
//
#include <hip/hip_runtime.h>
#include <hip/hip_bf16.h>

typedef __bf16 v16bf __attribute__((ext_vector_type(16)));
typedef float  v8f   __attribute__((ext_vector_type(8)));
typedef unsigned int u32x4 __attribute__((ext_vector_type(4)));

#define N_TOK  32768       // B*T
#define DIM    512         // D
#define M_EMB  1024        // M
#define ALPHA  (-5.0f)
#define BM     32          // rows per workgroup
#define XS     520         // padded LDS stride in halves (1040B rows: 16B-aligned runs, conflict-free)

// LDS: xf32(65536) xh(33280) xl(33280) x2s(128) rowMax(128) rowSinv(128)
//      wsRedF(1024) wsRedI(1024) colAcc(4096) => 138624 B
#define GEMM_SMEM_BYTES (BM*DIM*4 + BM*XS*2*2 + BM*4*3 + 8*BM*4*2 + M_EMB*4)

// workspace layout (bytes)
#define WS_E2_OFF     0
#define WS_X2_OFF     (WS_E2_OFF + M_EMB*4)
#define WS_AVGP_OFF   (WS_X2_OFF + N_TOK*4)
#define WS_COMMIT_OFF (WS_AVGP_OFF + M_EMB*4)
#define WS_COUNTS_OFF (WS_COMMIT_OFF + 4)
#define WS_IDX_OFF    (WS_COUNTS_OFF + M_EMB*4)
#define WS_EHS_OFF    (((WS_IDX_OFF + N_TOK*4) + 15) & ~15)   // 16B aligned bf16 planes
#define WS_ELS_OFF    (WS_EHS_OFF + M_EMB*DIM*2)

union V16 { u32x4 q[2]; v16bf v; };

// ---------------------------------------------------------------- row ||.||^2
__global__ __launch_bounds__(256) void gvq_rownorm2_kernel(
    const float* __restrict__ A, float* __restrict__ out, int rows) {
  int w = (blockIdx.x * blockDim.x + threadIdx.x) >> 5;
  int lane = threadIdx.x & 31;
  if (w >= rows) return;
  const float* r = A + (size_t)w * DIM;
  float s = 0.f;
  #pragma unroll
  for (int i = 0; i < DIM; i += 128) {
    float4 v = *(const float4*)(r + i + lane * 4);
    s += v.x * v.x + v.y * v.y + v.z * v.z + v.w * v.w;
  }
  #pragma unroll
  for (int off = 16; off >= 1; off >>= 1) s += __shfl_xor(s, off);
  if (lane == 0) out[w] = s;
}

// ------------------------------------- emb f32 -> planar bf16 hi/lo (once)
__global__ __launch_bounds__(256) void gvq_esplit_kernel(
    const float* __restrict__ emb, __bf16* __restrict__ eh,
    __bf16* __restrict__ el) {
  size_t i = ((size_t)blockIdx.x * 256 + threadIdx.x) * 2;   // float2 granule
  float2 v = *(const float2*)&emb[i];
  __bf16 h0 = (__bf16)v.x, h1 = (__bf16)v.y;
  eh[i] = h0;  eh[i + 1] = h1;
  el[i] = (__bf16)(v.x - (float)h0);
  el[i + 1] = (__bf16)(v.y - (float)h1);
}

// ---------------------------------------------------------------------- zero
__global__ void gvq_zero_kernel(float* avgp, int* counts, float* commit) {
  int g = blockIdx.x * blockDim.x + threadIdx.x;
  if (g < M_EMB) { avgp[g] = 0.f; counts[g] = 0; }
  if (g == 0) commit[0] = 0.f;
}

// ------------------------------------------- fused WMMA GEMM + row statistics
__global__ __launch_bounds__(256) void gvq_gemm_stats_kernel(
    const float* __restrict__ x, const __bf16* __restrict__ ehs,
    const __bf16* __restrict__ els, const float* __restrict__ gumbel,
    const float* __restrict__ e2, const float* __restrict__ x2,
    float* __restrict__ avgp, int* __restrict__ counts,
    int* __restrict__ idxout) {
  extern __shared__ char smem[];
  float*  xf32    = (float*)smem;                     // [BM][DIM] raw tile
  __bf16* xh      = (__bf16*)(smem + BM * DIM * 4);   // [BM][XS]
  __bf16* xl      = xh + BM * XS;                     // [BM][XS]
  float*  x2s     = (float*)(xl + BM * XS);           // [BM]
  float*  rowMaxS = x2s + BM;                         // [BM]
  float*  rowSinv = rowMaxS + BM;                     // [BM]
  float*  wsRedF  = rowSinv + BM;                     // [8][BM]
  int*    wsRedI  = (int*)(wsRedF + 8 * BM);          // [8][BM]
  float*  colAcc  = (float*)(wsRedI + 8 * BM);        // [M_EMB]

  const int t  = threadIdx.x;
  const int n0 = blockIdx.x * BM;

  // ---- async DMA the contiguous 64KB x tile into LDS (ASYNCcnt path) ----
  {
    const char* gsrc = (const char*)(x + (size_t)n0 * DIM);
    unsigned lbase = (unsigned)(size_t)xf32 + (unsigned)(t * 16);
    unsigned goff  = (unsigned)(t * 16);
    #pragma unroll
    for (int i = 0; i < 16; i++) {               // 256 thr * 16 * 16B = 64KB
      asm volatile("global_load_async_to_lds_b128 %0, %1, %2"
                   :: "v"(lbase + i * 4096), "v"(goff + i * 4096), "s"(gsrc)
                   : "memory");
    }
    asm volatile("s_wait_asynccnt 0x0" ::: "memory");
  }
  if (t < BM) x2s[t] = x2[n0 + t];
  __syncthreads();

  // ---- split x tile f32 -> bf16 hi/lo into padded LDS (once per block) ----
  for (int i = t; i < BM * DIM / 2; i += 256) {
    int r = i >> 8;                 // 256 float2 per row
    int c = (i & 255) * 2;
    float2 v = *(const float2*)&xf32[r * DIM + c];
    __bf16 h0 = (__bf16)v.x, h1 = (__bf16)v.y;
    xh[r * XS + c]     = h0;  xh[r * XS + c + 1] = h1;
    xl[r * XS + c]     = (__bf16)(v.x - (float)h0);
    xl[r * XS + c + 1] = (__bf16)(v.y - (float)h1);
  }
  __syncthreads();

  const int wave  = t >> 5;
  const int lane  = t & 31;
  const int cl    = lane & 15;         // A row / B col within tile
  const int hif   = lane >> 4;         // half-wave selector
  const int wbase = wave * 128;        // this wave's 128-column slice of M

  v8f acc[2][8] = {};                  // 2 row-tiles x 8 col-tiles

  // -------- K loop: 16 steps of K=32; 48 WMMA per step, zero cvt VALU ------
  #pragma unroll 1
  for (int k0 = 0; k0 < DIM; k0 += 32) {
    V16 ah[2], al[2];
    #pragma unroll
    for (int rt = 0; rt < 2; rt++) {   // ISA A-layout: two 16B half-runs
      const u32x4* ph = (const u32x4*)&xh[(rt * 16 + cl) * XS + k0 + 8 * hif];
      const u32x4* pl = (const u32x4*)&xl[(rt * 16 + cl) * XS + k0 + 8 * hif];
      ah[rt].q[0] = ph[0];  ah[rt].q[1] = ph[2];
      al[rt].q[0] = pl[0];  al[rt].q[1] = pl[2];
    }
    #pragma unroll
    for (int ct = 0; ct < 8; ct++) {
      size_t eo = (size_t)(wbase + ct * 16 + cl) * DIM + k0 + 8 * hif;
      const u32x4* peh = (const u32x4*)(ehs + eo);   // ISA B-layout: two 16B half-runs
      const u32x4* pel = (const u32x4*)(els + eo);
      __builtin_prefetch(peh + 4, 0, 0);             // next k-chunk
      V16 bh, bl;
      bh.q[0] = peh[0];  bh.q[1] = peh[2];
      bl.q[0] = pel[0];  bl.q[1] = pel[2];
      #pragma unroll
      for (int rt = 0; rt < 2; rt++) {               // hi*hi + hi*lo + lo*hi
        acc[rt][ct] = __builtin_amdgcn_wmma_f32_16x16x32_bf16(
            false, ah[rt].v, false, bh.v, (short)0, acc[rt][ct], false, false);
        acc[rt][ct] = __builtin_amdgcn_wmma_f32_16x16x32_bf16(
            false, ah[rt].v, false, bl.v, (short)0, acc[rt][ct], false, false);
        acc[rt][ct] = __builtin_amdgcn_wmma_f32_16x16x32_bf16(
            false, al[rt].v, false, bh.v, (short)0, acc[rt][ct], false, false);
      }
    }
  }

  // ---- dot -> dmap = ALPHA*(||e||^2 + ||x||^2 - 2 x.e), in registers ----
  float e2v[8];
  #pragma unroll
  for (int ct = 0; ct < 8; ct++) e2v[ct] = e2[wbase + ct * 16 + cl];
  #pragma unroll
  for (int rt = 0; rt < 2; rt++)
    #pragma unroll
    for (int ct = 0; ct < 8; ct++)
      #pragma unroll
      for (int g = 0; g < 8; g++) {
        int rl = rt * 16 + g + 8 * hif;              // C-matrix layout row
        acc[rt][ct][g] = ALPHA * (e2v[ct] + x2s[rl] - 2.0f * acc[rt][ct][g]);
      }

  // ---- row max: 16-lane shfl + cross-wave LDS ----
  #pragma unroll
  for (int rt = 0; rt < 2; rt++)
    #pragma unroll
    for (int g = 0; g < 8; g++) {
      float m = acc[rt][0][g];
      #pragma unroll
      for (int ct = 1; ct < 8; ct++) m = fmaxf(m, acc[rt][ct][g]);
      #pragma unroll
      for (int off = 8; off >= 1; off >>= 1) m = fmaxf(m, __shfl_xor(m, off));
      if (cl == 0) wsRedF[wave * BM + rt * 16 + g + 8 * hif] = m;
    }
  __syncthreads();
  if (t < BM) {
    float m = wsRedF[t];
    #pragma unroll
    for (int w = 1; w < 8; w++) m = fmaxf(m, wsRedF[w * BM + t]);
    rowMaxS[t] = m;
  }
  __syncthreads();

  // ---- row sum(exp) -> reciprocal ----
  #pragma unroll
  for (int rt = 0; rt < 2; rt++)
    #pragma unroll
    for (int g = 0; g < 8; g++) {
      int rl = rt * 16 + g + 8 * hif;
      float rm = rowMaxS[rl], s = 0.f;
      #pragma unroll
      for (int ct = 0; ct < 8; ct++) s += __expf(acc[rt][ct][g] - rm);
      #pragma unroll
      for (int off = 8; off >= 1; off >>= 1) s += __shfl_xor(s, off);
      if (cl == 0) wsRedF[wave * BM + rl] = s;
    }
  __syncthreads();
  if (t < BM) {
    float s = 0.f;
    #pragma unroll
    for (int w = 0; w < 8; w++) s += wsRedF[w * BM + t];
    rowSinv[t] = 1.0f / s;
  }
  __syncthreads();

  // ---- hard argmax(dmap) -> histogram ----
  #pragma unroll
  for (int rt = 0; rt < 2; rt++)
    #pragma unroll
    for (int g = 0; g < 8; g++) {
      float bv = acc[rt][0][g]; int bc = wbase + cl;
      #pragma unroll
      for (int ct = 1; ct < 8; ct++) {
        float v = acc[rt][ct][g]; int c = wbase + ct * 16 + cl;
        if (v > bv || (v == bv && c < bc)) { bv = v; bc = c; }
      }
      #pragma unroll
      for (int off = 8; off >= 1; off >>= 1) {
        float ov = __shfl_xor(bv, off); int oc = __shfl_xor(bc, off);
        if (ov > bv || (ov == bv && oc < bc)) { bv = ov; bc = oc; }
      }
      if (cl == 0) { int rl = rt * 16 + g + 8 * hif;
        wsRedF[wave * BM + rl] = bv; wsRedI[wave * BM + rl] = bc; }
    }
  __syncthreads();
  if (t < BM) {
    float bv = wsRedF[t]; int bc = wsRedI[t];
    #pragma unroll
    for (int w = 1; w < 8; w++) {
      float ov = wsRedF[w * BM + t]; int oc = wsRedI[w * BM + t];
      if (ov > bv || (ov == bv && oc < bc)) { bv = ov; bc = oc; }
    }
    atomicAdd(&counts[bc], 1);
  }
  __syncthreads();

  // ---- argmax(dmap + gumbel) -> selected code per row ----
  #pragma unroll
  for (int rt = 0; rt < 2; rt++)
    #pragma unroll
    for (int g = 0; g < 8; g++) {
      int rl = rt * 16 + g + 8 * hif;
      const float* grow = gumbel + (size_t)(n0 + rl) * M_EMB;
      float bv = -3.4e38f; int bc = 0;
      #pragma unroll
      for (int ct = 0; ct < 8; ct++) {
        int c = wbase + ct * 16 + cl;
        float v = acc[rt][ct][g] + grow[c];
        if (v > bv || (v == bv && c < bc)) { bv = v; bc = c; }
      }
      #pragma unroll
      for (int off = 8; off >= 1; off >>= 1) {
        float ov = __shfl_xor(bv, off); int oc = __shfl_xor(bc, off);
        if (ov > bv || (ov == bv && oc < bc)) { bv = ov; bc = oc; }
      }
      if (cl == 0) { wsRedF[wave * BM + rl] = bv; wsRedI[wave * BM + rl] = bc; }
    }
  __syncthreads();
  if (t < BM) {
    float bv = wsRedF[t]; int bc = wsRedI[t];
    #pragma unroll
    for (int w = 1; w < 8; w++) {
      float ov = wsRedF[w * BM + t]; int oc = wsRedI[w * BM + t];
      if (ov > bv || (ov == bv && oc < bc)) { bv = ov; bc = oc; }
    }
    idxout[n0 + t] = bc;
  }

  // ---- softmax column sums (avg_probs numerator) ----
  #pragma unroll
  for (int ct = 0; ct < 8; ct++) {
    float s = 0.f;
    #pragma unroll
    for (int rt = 0; rt < 2; rt++)
      #pragma unroll
      for (int g = 0; g < 8; g++) {
        int rl = rt * 16 + g + 8 * hif;
        s += __expf(acc[rt][ct][g] - rowMaxS[rl]) * rowSinv[rl];
      }
    s += __shfl_xor(s, 16);                         // merge half-waves
    if (hif == 0) colAcc[wbase + ct * 16 + cl] = s; // unique writer per col
  }
  __syncthreads();
  for (int j = t; j < M_EMB; j += 256) atomicAdd(&avgp[j], colAcc[j]);
}

// -------------------------------------- gather quantized + commitment loss
__global__ __launch_bounds__(256) void gvq_gather_kernel(
    const float* __restrict__ x, const float* __restrict__ emb,
    const int* __restrict__ idx, float* __restrict__ out,
    float* __restrict__ commit) {
  __shared__ float red[256];
  size_t i = (size_t)blockIdx.x * 256 + threadIdx.x;
  int n = (int)(i >> 9), c = (int)(i & 511);
  float q = emb[(size_t)idx[n] * DIM + c];
  out[i] = q;
  float d = x[i] - q;
  red[threadIdx.x] = d * d;
  __syncthreads();
  #pragma unroll
  for (int s = 128; s > 0; s >>= 1) {
    if (threadIdx.x < s) red[threadIdx.x] += red[threadIdx.x + s];
    __syncthreads();
  }
  if (threadIdx.x == 0) atomicAdd(commit, red[0]);
}

// ----------------------------------------------------------------- finalize
__global__ void gvq_finalize_kernel(const int* __restrict__ counts,
                                    const float* __restrict__ avgp,
                                    const float* __restrict__ commit,
                                    float* __restrict__ out) {
  __shared__ float r1[256], r2[256];
  float cp = 0.f, pp = 0.f;
  for (int c = threadIdx.x; c < M_EMB; c += 256) {
    float hp = (float)counts[c] * (1.0f / N_TOK);
    cp -= hp * log2f(hp + 1e-10f);
    float ap = avgp[c] * (1.0f / N_TOK);
    pp -= ap * log2f(ap + 1e-10f);
  }
  r1[threadIdx.x] = cp; r2[threadIdx.x] = pp;
  __syncthreads();
  for (int s = 128; s > 0; s >>= 1) {
    if (threadIdx.x < s) { r1[threadIdx.x] += r1[threadIdx.x + s];
                           r2[threadIdx.x] += r2[threadIdx.x + s]; }
    __syncthreads();
  }
  if (threadIdx.x == 0) {
    size_t base = (size_t)N_TOK * DIM;
    out[base + 0] = r1[0];
    out[base + 1] = r2[0];
    out[base + 2] = commit[0] / ((float)N_TOK * (float)DIM);
  }
}

extern "C" void kernel_launch(void* const* d_in, const int* in_sizes, int n_in,
                              void* d_out, int out_size, void* d_ws, size_t ws_size,
                              hipStream_t stream) {
  const float* x   = (const float*)d_in[0];  // [N,D]
  const float* emb = (const float*)d_in[1];  // [M,D]
  const float* gum = (const float*)d_in[2];  // [N,M]
  float* out = (float*)d_out;

  char* ws = (char*)d_ws;
  float*  e2     = (float*)(ws + WS_E2_OFF);
  float*  x2     = (float*)(ws + WS_X2_OFF);
  float*  avgp   = (float*)(ws + WS_AVGP_OFF);
  float*  commit = (float*)(ws + WS_COMMIT_OFF);
  int*    counts = (int*)(ws + WS_COUNTS_OFF);
  int*    idx    = (int*)(ws + WS_IDX_OFF);
  __bf16* ehs    = (__bf16*)(ws + WS_EHS_OFF);
  __bf16* els    = (__bf16*)(ws + WS_ELS_OFF);

  gvq_zero_kernel<<<(M_EMB + 255) / 256, 256, 0, stream>>>(avgp, counts, commit);
  gvq_esplit_kernel<<<M_EMB * DIM / 512, 256, 0, stream>>>(emb, ehs, els);
  gvq_rownorm2_kernel<<<M_EMB / 8, 256, 0, stream>>>(emb, e2, M_EMB);
  gvq_rownorm2_kernel<<<N_TOK / 8, 256, 0, stream>>>(x, x2, N_TOK);
  gvq_gemm_stats_kernel<<<N_TOK / BM, 256, GEMM_SMEM_BYTES, stream>>>(
      x, ehs, els, gum, e2, x2, avgp, counts, idx);
  gvq_gather_kernel<<<(N_TOK * DIM) / 256, 256, 0, stream>>>(x, emb, idx, out, commit);
  gvq_finalize_kernel<<<1, 256, 0, stream>>>(counts, avgp, commit, out);
}
